// LlamaAttention_10763188043892
// MI455X (gfx1250) — matrix operand order repaired and checked by reading, not compiled, not run on hardware
//
#include <hip/hip_runtime.h>

// Problem constants (match reference)
#define B_    2
#define S_    1024
#define HID_  4096
#define NH_   32
#define NKV_  8
#define HD_   128
#define G_    4
#define PAST_ 1024
#define L_    2048
#define QKVN  6144   // NH*HD + 2*NKV*HD

typedef __attribute__((ext_vector_type(16))) __bf16 bf16x16;
typedef __attribute__((ext_vector_type(8)))  __bf16 bf16x8;
typedef __attribute__((ext_vector_type(4)))  __bf16 bf16x4;
typedef __attribute__((ext_vector_type(8)))  float  f32x8;
typedef __attribute__((ext_vector_type(4)))  float  f32x4;

union ABf { bf16x16 v; bf16x8 h[2]; };

__device__ __forceinline__ bf16x4 cv4(f32x4 x) {
  return __builtin_convertvector(x, bf16x4);   // native v_cvt_pk_bf16_f32 path
}
__device__ __forceinline__ bf16x8 pack8(bf16x4 a, bf16x4 b) {
  return __builtin_shufflevector(a, b, 0, 1, 2, 3, 4, 5, 6, 7);
}

// ---------------------------------------------------------------------------
// fp32-in / bf16-WMMA / fp32-out GEMM.
// Block = 256 threads (8 waves). Workgroup tile 64(M) x 128(N), K-step 32.
// Wave w: wm=(w>>2)*32, wn=(w&3)*32 -> 32x32 wave tile = 4 WMMAs / K-step.
// ---------------------------------------------------------------------------
__global__ __launch_bounds__(256)
void gemm_bf16_wmma(const float* __restrict__ A, int lda,
                    const float* __restrict__ Bm, int ldb,
                    float* __restrict__ C, int ldc,
                    int M, int N, int K) {
  __shared__ __align__(16) __bf16 As[64][40];    // [m][k], padded to 80B rows
  __shared__ __align__(16) __bf16 Bs[128][40];   // [n][k] (transposed), padded

  const int tid  = threadIdx.x;
  const int lane = tid & 31;
  const int wave = tid >> 5;
  const int wm   = (wave >> 2) * 32;   // 0 / 32
  const int wn   = (wave & 3) * 32;    // 0..96
  const int m0   = blockIdx.y * 64;
  const int n0   = blockIdx.x * 128;

  const int ar  = tid >> 2;            // A stage: row 0..63
  const int ac  = (tid & 3) * 8;       //          k   0,8,16,24
  const int bkq = (tid & 7) * 4;       // B stage: k quad 0..28
  const int bn4 = (tid >> 3) * 4;      //          n quad 0..124

  const int lrow = lane & 15;
  const int hi   = lane >> 4;          // 0/1 lane-half

  f32x8 acc[2][2];
#pragma unroll
  for (int mi = 0; mi < 2; ++mi)
#pragma unroll
    for (int ni = 0; ni < 2; ++ni)
      acc[mi][ni] = (f32x8){0.f,0.f,0.f,0.f,0.f,0.f,0.f,0.f};

  for (int k0 = 0; k0 < K; k0 += 32) {
    // ---- stage A tile: 8 contiguous k per thread -> one 16B LDS store ----
    {
      const float* ap = A + (size_t)(m0 + ar) * lda + k0 + ac;
      const bf16x8 h = pack8(cv4(*(const f32x4*)ap), cv4(*(const f32x4*)(ap + 4)));
      *(bf16x8*)&As[ar][ac] = h;
    }
    // ---- stage B tile transposed: 4k x 4n micro-tile, packed b64 stores ----
    {
      const float* bp = Bm + (size_t)(k0 + bkq) * ldb + n0 + bn4;
      const bf16x4 h0 = cv4(*(const f32x4*)bp);
      const bf16x4 h1 = cv4(*(const f32x4*)(bp + ldb));
      const bf16x4 h2 = cv4(*(const f32x4*)(bp + 2 * (size_t)ldb));
      const bf16x4 h3 = cv4(*(const f32x4*)(bp + 3 * (size_t)ldb));
#pragma unroll
      for (int j = 0; j < 4; ++j) {
        bf16x4 col = {h0[j], h1[j], h2[j], h3[j]};
        *(bf16x4*)&Bs[bn4 + j][bkq] = col;     // k contiguous -> 8B store
      }
    }
    // prefetch next K tile (global_prefetch_b8)
    if (k0 + 32 < K) {
      __builtin_prefetch(A + (size_t)(m0 + ar) * lda + k0 + 32 + ac, 0, 1);
      __builtin_prefetch(Bm + (size_t)(k0 + 32 + bkq) * ldb + n0 + bn4, 0, 1);
    }
    __syncthreads();

    // ---- fragments + 4 WMMAs ----
    ABf a[2], b[2];
#pragma unroll
    for (int mi = 0; mi < 2; ++mi) {
      a[mi].h[0] = *(const bf16x8*)&As[wm + mi * 16 + lrow][hi * 8];
      a[mi].h[1] = *(const bf16x8*)&As[wm + mi * 16 + lrow][hi * 8 + 16];
    }
#pragma unroll
    for (int ni = 0; ni < 2; ++ni) {
      b[ni].h[0] = *(const bf16x8*)&Bs[wn + ni * 16 + lrow][hi * 8];
      b[ni].h[1] = *(const bf16x8*)&Bs[wn + ni * 16 + lrow][hi * 8 + 16];
    }
#pragma unroll
    for (int mi = 0; mi < 2; ++mi)
#pragma unroll
      for (int ni = 0; ni < 2; ++ni)
        acc[mi][ni] = __builtin_amdgcn_wmma_f32_16x16x32_bf16(
            false, a[mi].v, false, b[ni].v, (short)0, acc[mi][ni], false, false);
    __syncthreads();
  }

  // ---- store C (VGPR r -> row r + 8*hi, col = lane&15) ----
#pragma unroll
  for (int mi = 0; mi < 2; ++mi)
#pragma unroll
    for (int ni = 0; ni < 2; ++ni) {
      float* cp = C + (size_t)(m0 + wm + mi * 16 + hi * 8) * ldc +
                  n0 + wn + ni * 16 + lrow;
#pragma unroll
      for (int r = 0; r < 8; ++r)
        cp[(size_t)r * ldc] = acc[mi][ni][r];
    }
}

// ---------------------------------------------------------------------------
// RoPE: rope Q in place in the fused QKV workspace; rope K and copy V into
// the fresh-KV scratch buffer [B*S][2][NKV][HD] (kv_cache stays read-only).
// ---------------------------------------------------------------------------
__global__ __launch_bounds__(128)
void rope_scatter(float* __restrict__ qkv,          // [B*S][QKVN]
                  float* __restrict__ newkv,        // [B*S][2][NKV][HD]
                  const float* __restrict__ rope,   // [MAXPOS][2][HD]
                  const int* __restrict__ pos_off)  // [B]
{
  const int t = blockIdx.x;
  const int b = t / S_;
  const int s = t - b * S_;
  const int pos = pos_off[b] + s;
  const float* cosr = rope + ((size_t)pos * 2 + 0) * HD_;
  const float* sinr = rope + ((size_t)pos * 2 + 1) * HD_;
  float* qrow = qkv + (size_t)t * QKVN;

  for (int idx = threadIdx.x; idx < NH_ * 64; idx += blockDim.x) {
    const int h = idx >> 6, d = idx & 63;
    float* qh = qrow + h * HD_;
    const float x1 = qh[d], x2 = qh[d + 64];
    qh[d]      = x1 * cosr[d]      - x2 * sinr[d];
    qh[d + 64] = x2 * cosr[d + 64] + x1 * sinr[d + 64];
  }
  const float* krow = qrow + HID_;
  float* kout = newkv + (size_t)t * 2 * NKV_ * HD_;
  for (int idx = threadIdx.x; idx < NKV_ * 64; idx += blockDim.x) {
    const int h = idx >> 6, d = idx & 63;
    const float* kh = krow + h * HD_;
    const float x1 = kh[d], x2 = kh[d + 64];
    kout[h * HD_ + d]      = x1 * cosr[d]      - x2 * sinr[d];
    kout[h * HD_ + d + 64] = x2 * cosr[d + 64] + x1 * sinr[d + 64];
  }
  const float* vrow = qrow + HID_ + NKV_ * HD_;
  float* vout = kout + NKV_ * HD_;
  for (int idx = threadIdx.x; idx < NKV_ * HD_; idx += blockDim.x)
    vout[idx] = vrow[idx];
}

// ---------------------------------------------------------------------------
// Flash attention: 1 wave per (b, head, 16-query tile). Online softmax,
// 32-key chunks: QK^T = 8 WMMAs, PV = 8 WMMAs per chunk.
// ---------------------------------------------------------------------------
__global__ __launch_bounds__(32)
void attn_fwd(const float* __restrict__ qkv,      // rope'd Q in cols [0,4096)
              const float* __restrict__ kvcache,  // [SLOTS][2][NKV][HD]
              const float* __restrict__ newkv,    // [B*S][2][NKV][HD]
              const int* __restrict__ kv_lut,     // [B][L]
              const int* __restrict__ pos_off,    // [B]
              float* __restrict__ attn_out)       // [B*S][NH*HD]
{
  __shared__ __align__(16) __bf16 Ks[32][HD_];  // keys  [key][dim]
  __shared__ __align__(16) __bf16 Vt[HD_][32];  // vals  [dim][key]
  __shared__ __align__(16) __bf16 Ps[16][32];   // probs [row][key]

  const int qt   = blockIdx.x;
  const int h    = blockIdx.y;
  const int b    = blockIdx.z;
  const int kvh  = h / G_;
  const int lane = threadIdx.x & 31;
  const int lrow = lane & 15;
  const int hi   = lane >> 4;
  const int posb = pos_off[b];

  // ---- Q fragments (4 x 16x32 bf16, converted from fp32 workspace) ----
  ABf qf[4];
  {
    const float* qp = qkv + (size_t)(b * S_ + qt * 16 + lrow) * QKVN + h * HD_;
#pragma unroll
    for (int f = 0; f < 4; ++f) {
      const float* p0 = qp + f * 32 + hi * 8;
      qf[f].h[0] = pack8(cv4(*(const f32x4*)p0),        cv4(*(const f32x4*)(p0 + 4)));
      qf[f].h[1] = pack8(cv4(*(const f32x4*)(p0 + 16)), cv4(*(const f32x4*)(p0 + 20)));
    }
  }

  f32x8 o[8];
#pragma unroll
  for (int g = 0; g < 8; ++g) o[g] = (f32x8){0.f,0.f,0.f,0.f,0.f,0.f,0.f,0.f};
  float mrun[8], lrun[8];
#pragma unroll
  for (int r = 0; r < 8; ++r) { mrun[r] = -1e30f; lrun[r] = 0.f; }

  const float scale = 0.08838834764831845f;  // 1/sqrt(128)
  const int lmaxp = posb + qt * 16 + 15;
  int nch = lmaxp / 32 + 1;
  if (nch > L_ / 32) nch = L_ / 32;

  for (int ch = 0; ch < nch; ++ch) {
    // ---- stage 32 keys + values to LDS (one key per lane) ----
    {
      const int lglob = ch * 32 + lane;
      const float* src;
      if (lglob < PAST_) {
        const int slot = kv_lut[b * L_ + lglob];
        src = kvcache + (size_t)slot * 2 * (NKV_ * HD_) + kvh * HD_;
      } else {
        src = newkv + (size_t)(b * S_ + (lglob - PAST_)) * 2 * (NKV_ * HD_) + kvh * HD_;
      }
      const float* vsrc = src + NKV_ * HD_;
#pragma unroll
      for (int d = 0; d < HD_; d += 4) {
        *(bf16x4*)&Ks[lane][d] = cv4(*(const f32x4*)(src + d));  // 8B store
        const bf16x4 hv = cv4(*(const f32x4*)(vsrc + d));
        Vt[d + 0][lane] = hv[0];
        Vt[d + 1][lane] = hv[1];
        Vt[d + 2][lane] = hv[2];
        Vt[d + 3][lane] = hv[3];
      }
    }
    __syncthreads();

    // ---- scores: Q(16x128) . K^T(128x32) as two 16x16 C tiles ----
    f32x8 sc0 = {0.f,0.f,0.f,0.f,0.f,0.f,0.f,0.f};
    f32x8 sc1 = {0.f,0.f,0.f,0.f,0.f,0.f,0.f,0.f};
#pragma unroll
    for (int f = 0; f < 4; ++f) {
      ABf b0, b1;
      b0.h[0] = *(const bf16x8*)&Ks[lrow][f * 32 + hi * 8];
      b0.h[1] = *(const bf16x8*)&Ks[lrow][f * 32 + hi * 8 + 16];
      b1.h[0] = *(const bf16x8*)&Ks[16 + lrow][f * 32 + hi * 8];
      b1.h[1] = *(const bf16x8*)&Ks[16 + lrow][f * 32 + hi * 8 + 16];
      sc0 = __builtin_amdgcn_wmma_f32_16x16x32_bf16(false, qf[f].v, false, b0.v,
                                                    (short)0, sc0, false, false);
      sc1 = __builtin_amdgcn_wmma_f32_16x16x32_bf16(false, qf[f].v, false, b1.v,
                                                    (short)0, sc1, false, false);
    }

    // ---- causal mask + online softmax (row reductions over 16-lane groups) ----
#pragma unroll
    for (int r = 0; r < 8; ++r) {
      const int row  = r + hi * 8;
      const int qpos = posb + qt * 16 + row;
      const int kp0  = ch * 32 + lrow;
      float s0 = (kp0      <= qpos) ? sc0[r] * scale : -1e30f;
      float s1 = (kp0 + 16 <= qpos) ? sc1[r] * scale : -1e30f;
      float mloc = fmaxf(s0, s1);
#pragma unroll
      for (int off = 1; off < 16; off <<= 1)
        mloc = fmaxf(mloc, __shfl_xor(mloc, off, 32));
      const float mnew = fmaxf(mrun[r], mloc);
      const float corr = __expf(mrun[r] - mnew);
      const float p0 = __expf(s0 - mnew);
      const float p1 = __expf(s1 - mnew);
      float psum = p0 + p1;
#pragma unroll
      for (int off = 1; off < 16; off <<= 1)
        psum += __shfl_xor(psum, off, 32);
      lrun[r] = lrun[r] * corr + psum;
      mrun[r] = mnew;
#pragma unroll
      for (int g = 0; g < 8; ++g) o[g][r] *= corr;
      Ps[row][lrow]      = (__bf16)p0;
      Ps[row][lrow + 16] = (__bf16)p1;
    }
    __syncthreads();

    // ---- P(16x32) . V(32x128): 8 WMMAs over the dim groups ----
    ABf pf;
    pf.h[0] = *(const bf16x8*)&Ps[lrow][hi * 8];
    pf.h[1] = *(const bf16x8*)&Ps[lrow][hi * 8 + 16];
#pragma unroll
    for (int g = 0; g < 8; ++g) {
      ABf vb;
      vb.h[0] = *(const bf16x8*)&Vt[g * 16 + lrow][hi * 8];
      vb.h[1] = *(const bf16x8*)&Vt[g * 16 + lrow][hi * 8 + 16];
      o[g] = __builtin_amdgcn_wmma_f32_16x16x32_bf16(false, pf.v, false, vb.v,
                                                     (short)0, o[g], false, false);
    }
    __syncthreads();
  }

  // ---- finalize: divide by row sums, write [token][h*128 + d] fp32 ----
  float* orow = attn_out + (size_t)(b * S_ + qt * 16 + hi * 8) * (NH_ * HD_) + h * HD_;
#pragma unroll
  for (int r = 0; r < 8; ++r) {
    const float inv = 1.0f / lrun[r];
#pragma unroll
    for (int g = 0; g < 8; ++g)
      orow[(size_t)r * (NH_ * HD_) + g * 16 + lrow] = o[g][r] * inv;
  }
}

// ---------------------------------------------------------------------------
extern "C" void kernel_launch(void* const* d_in, const int* in_sizes, int n_in,
                              void* d_out, int out_size, void* d_ws, size_t ws_size,
                              hipStream_t stream) {
  (void)in_sizes; (void)n_in; (void)out_size; (void)ws_size;

  const float* hs   = (const float*)d_in[0];  // hidden_states [B,S,HID]
  const float* kvc  = (const float*)d_in[1];  // kv_cache [SLOTS,2,NKV,HD] (read-only)
  const float* rope = (const float*)d_in[2];  // rope_cache [MAXPOS,2,HD]
  const float* Wq   = (const float*)d_in[3];  // [HID, NH*HD]
  const float* Wk   = (const float*)d_in[4];  // [HID, NKV*HD]
  const float* Wv   = (const float*)d_in[5];  // [HID, NKV*HD]
  const float* Wo   = (const float*)d_in[6];  // [NH*HD, HID]
  const int*   pos  = (const int*)d_in[7];    // [B]
  const int*   lutd = (const int*)d_in[8];    (void)lutd;
  const int*   lut  = (const int*)d_in[9];    // kv_lut [B,L]
  float* out = (float*)d_out;

  const int M = B_ * S_;  // 2048 tokens

  // workspace carve-up
  float* qkv   = (float*)d_ws;                               // [M][QKVN]        48 MB
  float* newkv = qkv + (size_t)M * QKVN;                     // [M][2][NKV][HD]  16 MB
  float* attn  = newkv + (size_t)M * 2 * NKV_ * HD_;         // [M][NH*HD]       32 MB

  const dim3 blk(256);
  // QKV projections (fused output buffer, ldc = QKVN)
  gemm_bf16_wmma<<<dim3(HID_ / 128, M / 64), blk, 0, stream>>>(
      hs, HID_, Wq, HID_, qkv, QKVN, M, HID_, HID_);
  gemm_bf16_wmma<<<dim3((NKV_ * HD_) / 128, M / 64), blk, 0, stream>>>(
      hs, HID_, Wk, NKV_ * HD_, qkv + HID_, QKVN, M, NKV_ * HD_, HID_);
  gemm_bf16_wmma<<<dim3((NKV_ * HD_) / 128, M / 64), blk, 0, stream>>>(
      hs, HID_, Wv, NKV_ * HD_, qkv + HID_ + NKV_ * HD_, QKVN, M, NKV_ * HD_, HID_);

  // RoPE + fresh-KV scatter
  rope_scatter<<<M, 128, 0, stream>>>(qkv, newkv, rope, pos);

  // Flash attention
  attn_fwd<<<dim3(S_ / 16, NH_, B_), 32, 0, stream>>>(qkv, kvc, newkv, lut, pos, attn);

  // Output projection
  gemm_bf16_wmma<<<dim3(HID_ / 128, M / 64), blk, 0, stream>>>(
      attn, HID_, Wo, HID_, out, HID_, M, HID_, HID_);
}
